// protoLinear_15659450761902
// MI455X (gfx1250) — compile-verified, matrix operand order checked by reading, not run on hardware
//
#include <hip/hip_runtime.h>

#define INDIM  1024
#define OUTDIM 2048
#define BATCH  8192

typedef unsigned short u16;
typedef __attribute__((ext_vector_type(8)))  u16    u16x8;
typedef __attribute__((ext_vector_type(16))) u16    u16x16;
typedef __attribute__((ext_vector_type(16))) __bf16 bf16x16;
typedef __attribute__((ext_vector_type(8)))  float  f32x8;

// Split f32 into bf16 hi (RNE) + bf16 lo (RNE of remainder).
static __device__ __forceinline__ void split_bf16(float f, u16& hi, u16& lo) {
    unsigned u  = __builtin_bit_cast(unsigned, f);
    unsigned rh = u + 0x7FFFu + ((u >> 16) & 1u);
    hi = (u16)(rh >> 16);
    float hf  = __builtin_bit_cast(float, (unsigned)hi << 16);
    float rem = f - hf;
    unsigned ur = __builtin_bit_cast(unsigned, rem);
    unsigned rl = ur + 0x7FFFu + ((ur >> 16) & 1u);
    lo = (u16)(rl >> 16);
}

// Build a 16xbf16 fragment from two contiguous 16-byte LDS chunks.
static __device__ __forceinline__ bf16x16 load_frag(const u16* base, int off1, int off2) {
    u16x8 a = *(const u16x8*)(base + off1);
    u16x8 b = *(const u16x8*)(base + off2);
    u16x16 c = __builtin_shufflevector(a, b, 0,1,2,3,4,5,6,7,8,9,10,11,12,13,14,15);
    return __builtin_bit_cast(bf16x16, c);
}

// CDNA5 async memory->LDS copy, 16 B per lane, tracked by ASYNCcnt.
// LDS operand = low 32 bits of the generic pointer (flat->LDS keeps addr[31:0]).
static __device__ __forceinline__ void async_copy16(const u16* g, u16* l) {
    asm volatile("global_load_async_to_lds_b128 %0, %1, off"
                 :
                 : "v"((unsigned)(unsigned long long)l), "v"(g)
                 : "memory");
}
static __device__ __forceinline__ void wait_asynccnt0() {
    asm volatile("s_wait_asynccnt 0x0" ::: "memory");
}

// ---------- prologue: split X into bf16 hi/lo + row norms ----------
__global__ __launch_bounds__(256) void xsplit_kernel(const float* __restrict__ X,
                                                     u16* __restrict__ Xhi,
                                                     u16* __restrict__ Xlo,
                                                     float* __restrict__ xsq) {
    const int wave = threadIdx.x >> 5, lane = threadIdx.x & 31;
    const int row  = blockIdx.x * 8 + wave;
    const size_t rb = (size_t)row * INDIM;
    float s = 0.f;
#pragma unroll
    for (int i = 0; i < 8; ++i) {
        const int col = lane * 4 + i * 128;
        float4 v = *(const float4*)(X + rb + col);
        s += v.x * v.x + v.y * v.y + v.z * v.z + v.w * v.w;
        union { u16 e[4]; uint2 u; } ph, pl;
        split_bf16(v.x, ph.e[0], pl.e[0]);
        split_bf16(v.y, ph.e[1], pl.e[1]);
        split_bf16(v.z, ph.e[2], pl.e[2]);
        split_bf16(v.w, ph.e[3], pl.e[3]);
        *(uint2*)(Xhi + rb + col) = ph.u;
        *(uint2*)(Xlo + rb + col) = pl.u;
    }
#pragma unroll
    for (int off = 16; off > 0; off >>= 1) s += __shfl_xor(s, off, 32);
    if (lane == 0) xsq[row] = s;
}

// ---------- prologue: row norms only (fallback path) ----------
__global__ __launch_bounds__(256) void xsq_kernel(const float* __restrict__ X,
                                                  float* __restrict__ xsq) {
    const int wave = threadIdx.x >> 5, lane = threadIdx.x & 31;
    const int row  = blockIdx.x * 8 + wave;
    const float* xr = X + (size_t)row * INDIM;
    float s = 0.f;
#pragma unroll
    for (int i = 0; i < 8; ++i) {
        float4 v = *(const float4*)(xr + lane * 4 + i * 128);
        s += v.x * v.x + v.y * v.y + v.z * v.z + v.w * v.w;
    }
#pragma unroll
    for (int off = 16; off > 0; off >>= 1) s += __shfl_xor(s, off, 32);
    if (lane == 0) xsq[row] = s;
}

// ---------- prologue: column norms of P ----------
__global__ __launch_bounds__(256) void psq_kernel(const float* __restrict__ P,
                                                  float* __restrict__ psq) {
    const int o = blockIdx.x * 256 + threadIdx.x;
    float s = 0.f;
    for (int k = 0; k < INDIM; ++k) {
        float p = P[(size_t)k * OUTDIM + o];
        s += p * p;
    }
    psq[o] = s;
}

// ---------- prologue: split + transpose P -> Pt[n][k] bf16 hi/lo ----------
#define TP 72  // LDS pitch in u16 (144 B, 16-B aligned)
__global__ __launch_bounds__(256) void psplit_kernel(const float* __restrict__ P,
                                                     u16* __restrict__ Pthi,
                                                     u16* __restrict__ Ptlo) {
    __shared__ u16 th[64 * TP];
    __shared__ u16 tl[64 * TP];
    const int k0 = blockIdx.x * 64;
    const int n0 = blockIdx.y * 64;
    const int t  = threadIdx.x;
#pragma unroll
    for (int i = 0; i < 4; ++i) {
        const int q  = t + i * 256;
        const int k  = q >> 4;
        const int n4 = (q & 15) * 4;
        float4 v = *(const float4*)(P + (size_t)(k0 + k) * OUTDIM + n0 + n4);
        u16 hh, ll;
        split_bf16(v.x, hh, ll); th[(n4 + 0) * TP + k] = hh; tl[(n4 + 0) * TP + k] = ll;
        split_bf16(v.y, hh, ll); th[(n4 + 1) * TP + k] = hh; tl[(n4 + 1) * TP + k] = ll;
        split_bf16(v.z, hh, ll); th[(n4 + 2) * TP + k] = hh; tl[(n4 + 2) * TP + k] = ll;
        split_bf16(v.w, hh, ll); th[(n4 + 3) * TP + k] = hh; tl[(n4 + 3) * TP + k] = ll;
    }
    __syncthreads();
#pragma unroll
    for (int i = 0; i < 2; ++i) {
        const int q = t + i * 256;
        const int n = q >> 3;
        const int c = (q & 7) * 8;
        const size_t g = (size_t)(n0 + n) * INDIM + k0 + c;
        *(uint4*)(Pthi + g) = *(const uint4*)(th + n * TP + c);
        *(uint4*)(Ptlo + g) = *(const uint4*)(tl + n * TP + c);
    }
}

// ---------- main fused GEMM: async double-buffered LDS staging ----------
// Block tile 128x128, BK=32, 256 threads = 8 waves (2x4 grid, 64x32 each).
#define LDP 40                 // LDS row pitch in u16 (80 B)
#define TSZ (128 * LDP)        // one array
#define BSZ (4 * TSZ)          // one stage buffer (Ahi,Alo,Bhi,Blo)

__global__ __launch_bounds__(256, 2) void proto_gemm_split(const u16* __restrict__ Xhi,
                                                           const u16* __restrict__ Xlo,
                                                           const u16* __restrict__ Pthi,
                                                           const u16* __restrict__ Ptlo,
                                                           const float* __restrict__ xsq,
                                                           const float* __restrict__ psq,
                                                           float* __restrict__ out) {
    __shared__ u16 smem[2 * BSZ];   // 80 KB: double-buffered tiles

    const int t      = threadIdx.x;
    const int blockM = blockIdx.x * 128;
    const int blockN = blockIdx.y * 128;
    const int wave   = t >> 5;
    const int lane   = t & 31;
    const int wm     = wave >> 2;   // 0..1  (M)
    const int wn     = wave & 3;    // 0..3  (N)
    const int r      = lane & 15;
    const int h      = lane >> 4;
    const int k1     = h * 8;       // K elems 0 or 8
    const int k2     = k1 + 16;     // K elems 16 or 24

    f32x8 acc[4][2];
#pragma unroll
    for (int mt = 0; mt < 4; ++mt)
#pragma unroll
        for (int nt = 0; nt < 2; ++nt) acc[mt][nt] = (f32x8)0.0f;

    // Per-thread staging coords: 512 16-B chunks per array, 2 per thread.
    const int srow0 = t >> 2;
    const int sc    = (t & 3) * 8;
    const int NK    = INDIM / 32;

    // Issue async loads for one K tile into buffer b.
    auto stage = [&](int kt, int b) {
        u16* Ah = smem + b * BSZ;
        u16* Al = Ah + TSZ;
        u16* Bh = Al + TSZ;
        u16* Bl = Bh + TSZ;
        const int k0 = kt * 32;
#pragma unroll
        for (int i = 0; i < 2; ++i) {
            const int row = srow0 + i * 64;
            const size_t ga = (size_t)(blockM + row) * INDIM + k0 + sc;
            const size_t gb = (size_t)(blockN + row) * INDIM + k0 + sc;
            const int lo = row * LDP + sc;
            async_copy16(Xhi  + ga, Ah + lo);
            async_copy16(Xlo  + ga, Al + lo);
            async_copy16(Pthi + gb, Bh + lo);
            async_copy16(Ptlo + gb, Bl + lo);
        }
    };

    stage(0, 0);

    for (int kt = 0; kt < NK; ++kt) {
        wait_asynccnt0();       // this wave's DMA for tile kt has landed in LDS
        __syncthreads();        // whole block's tile kt visible; prev buffer free
        if (kt + 1 < NK) stage(kt + 1, (kt + 1) & 1);

        const u16* Ah = smem + (kt & 1) * BSZ;
        const u16* Al = Ah + TSZ;
        const u16* Bh = Al + TSZ;
        const u16* Bl = Bh + TSZ;

        bf16x16 bhv[2], blv[2];
#pragma unroll
        for (int nt = 0; nt < 2; ++nt) {
            const int col = wn * 32 + nt * 16 + r;
            bhv[nt] = load_frag(Bh + col * LDP, k1, k2);
            blv[nt] = load_frag(Bl + col * LDP, k1, k2);
        }
#pragma unroll
        for (int mt = 0; mt < 4; ++mt) {
            const int rowl = wm * 64 + mt * 16 + r;
            bf16x16 ah = load_frag(Ah + rowl * LDP, k1, k2);
            bf16x16 al = load_frag(Al + rowl * LDP, k1, k2);
#pragma unroll
            for (int nt = 0; nt < 2; ++nt) {
                acc[mt][nt] = __builtin_amdgcn_wmma_f32_16x16x32_bf16(
                    false, ah, false, bhv[nt], (short)0, acc[mt][nt], false, false);
                acc[mt][nt] = __builtin_amdgcn_wmma_f32_16x16x32_bf16(
                    false, ah, false, blv[nt], (short)0, acc[mt][nt], false, false);
                acc[mt][nt] = __builtin_amdgcn_wmma_f32_16x16x32_bf16(
                    false, al, false, bhv[nt], (short)0, acc[mt][nt], false, false);
            }
        }
    }

    float psv[2];
#pragma unroll
    for (int nt = 0; nt < 2; ++nt) psv[nt] = psq[blockN + wn * 32 + nt * 16 + r];
#pragma unroll
    for (int mt = 0; mt < 4; ++mt) {
#pragma unroll
        for (int j = 0; j < 8; ++j) {
            const int row = blockM + wm * 64 + mt * 16 + h * 8 + j;
            const float xs = xsq[row];
#pragma unroll
            for (int nt = 0; nt < 2; ++nt) {
                const int col = blockN + wn * 32 + nt * 16 + r;
                out[(size_t)row * OUTDIM + col] = 2.0f * acc[mt][nt][j] - xs - psv[nt];
            }
        }
    }
}

// ---------- fallback GEMM (in-loop split), used if d_ws is too small ----------
__global__ __launch_bounds__(256, 2) void proto_gemm_fused(const float* __restrict__ X,
                                                           const float* __restrict__ P,
                                                           const float* __restrict__ xsq,
                                                           const float* __restrict__ psq,
                                                           float* __restrict__ out) {
    __shared__ u16 Ahi[128 * LDP];
    __shared__ u16 Alo[128 * LDP];
    __shared__ u16 Bhi[128 * LDP];
    __shared__ u16 Blo[128 * LDP];

    const int t      = threadIdx.x;
    const int blockM = blockIdx.x * 128;
    const int blockN = blockIdx.y * 128;
    const int wave   = t >> 5;
    const int lane   = t & 31;
    const int wm     = wave >> 2;
    const int wn     = wave & 3;
    const int r      = lane & 15;
    const int h      = lane >> 4;
    const int k1     = h * 8;
    const int k2     = k1 + 16;

    f32x8 acc[4][2];
#pragma unroll
    for (int mt = 0; mt < 4; ++mt)
#pragma unroll
        for (int nt = 0; nt < 2; ++nt) acc[mt][nt] = (f32x8)0.0f;

    for (int kt = 0; kt < INDIM / 32; ++kt) {
        const int k0 = kt * 32;
#pragma unroll
        for (int i = 0; i < 4; ++i) {
            const int q   = t + i * 256;
            const int row = q >> 3;
            const int kc  = (q & 7) * 4;
            const float4 v = *(const float4*)(X + (size_t)(blockM + row) * INDIM + k0 + kc);
            union { u16 e[4]; uint2 u; } ph, pl;
            split_bf16(v.x, ph.e[0], pl.e[0]);
            split_bf16(v.y, ph.e[1], pl.e[1]);
            split_bf16(v.z, ph.e[2], pl.e[2]);
            split_bf16(v.w, ph.e[3], pl.e[3]);
            *(uint2*)(Ahi + row * LDP + kc) = ph.u;
            *(uint2*)(Alo + row * LDP + kc) = pl.u;
        }
#pragma unroll
        for (int i = 0; i < 4; ++i) {
            const int q  = t + i * 256;
            const int kk = q >> 5;
            const int nc = (q & 31) * 4;
            const float4 v = *(const float4*)(P + (size_t)(k0 + kk) * OUTDIM + blockN + nc);
            u16 hh, ll;
            split_bf16(v.x, hh, ll); Bhi[(nc + 0) * LDP + kk] = hh; Blo[(nc + 0) * LDP + kk] = ll;
            split_bf16(v.y, hh, ll); Bhi[(nc + 1) * LDP + kk] = hh; Blo[(nc + 1) * LDP + kk] = ll;
            split_bf16(v.z, hh, ll); Bhi[(nc + 2) * LDP + kk] = hh; Blo[(nc + 2) * LDP + kk] = ll;
            split_bf16(v.w, hh, ll); Bhi[(nc + 3) * LDP + kk] = hh; Blo[(nc + 3) * LDP + kk] = ll;
        }
        __syncthreads();

        bf16x16 bh[2], bl[2];
#pragma unroll
        for (int nt = 0; nt < 2; ++nt) {
            const int col = wn * 32 + nt * 16 + r;
            bh[nt] = load_frag(Bhi + col * LDP, k1, k2);
            bl[nt] = load_frag(Blo + col * LDP, k1, k2);
        }
#pragma unroll
        for (int mt = 0; mt < 4; ++mt) {
            const int rowl = wm * 64 + mt * 16 + r;
            bf16x16 ah = load_frag(Ahi + rowl * LDP, k1, k2);
            bf16x16 al = load_frag(Alo + rowl * LDP, k1, k2);
#pragma unroll
            for (int nt = 0; nt < 2; ++nt) {
                acc[mt][nt] = __builtin_amdgcn_wmma_f32_16x16x32_bf16(
                    false, ah, false, bh[nt], (short)0, acc[mt][nt], false, false);
                acc[mt][nt] = __builtin_amdgcn_wmma_f32_16x16x32_bf16(
                    false, ah, false, bl[nt], (short)0, acc[mt][nt], false, false);
                acc[mt][nt] = __builtin_amdgcn_wmma_f32_16x16x32_bf16(
                    false, al, false, bh[nt], (short)0, acc[mt][nt], false, false);
            }
        }
        __syncthreads();
    }

    float psv[2];
#pragma unroll
    for (int nt = 0; nt < 2; ++nt) psv[nt] = psq[blockN + wn * 32 + nt * 16 + r];
#pragma unroll
    for (int mt = 0; mt < 4; ++mt) {
#pragma unroll
        for (int j = 0; j < 8; ++j) {
            const int row = blockM + wm * 64 + mt * 16 + h * 8 + j;
            const float xs = xsq[row];
#pragma unroll
            for (int nt = 0; nt < 2; ++nt) {
                const int col = blockN + wn * 32 + nt * 16 + r;
                out[(size_t)row * OUTDIM + col] = 2.0f * acc[mt][nt][j] - xs - psv[nt];
            }
        }
    }
}

extern "C" void kernel_launch(void* const* d_in, const int* in_sizes, int n_in,
                              void* d_out, int out_size, void* d_ws, size_t ws_size,
                              hipStream_t stream) {
    const float* X = (const float*)d_in[0];
    const float* P = (const float*)d_in[1];
    float* out = (float*)d_out;

    float* xsq = (float*)d_ws;                 // BATCH floats
    float* psq = xsq + BATCH;                  // OUTDIM floats
    const size_t hdr = (size_t)(BATCH + OUTDIM) * sizeof(float);          // 40 KB
    const size_t nx  = (size_t)BATCH * INDIM;
    const size_t np  = (size_t)OUTDIM * INDIM;
    const size_t need = hdr + 2 * (nx + np) * sizeof(u16);                // ~40 MB

    if (ws_size >= need) {
        u16* Xhi  = (u16*)((char*)d_ws + hdr);
        u16* Xlo  = Xhi + nx;
        u16* Pthi = Xlo + nx;
        u16* Ptlo = Pthi + np;
        xsplit_kernel<<<BATCH / 8, 256, 0, stream>>>(X, Xhi, Xlo, xsq);
        psq_kernel<<<OUTDIM / 256, 256, 0, stream>>>(P, psq);
        psplit_kernel<<<dim3(INDIM / 64, OUTDIM / 64), 256, 0, stream>>>(P, Pthi, Ptlo);
        proto_gemm_split<<<dim3(BATCH / 128, OUTDIM / 128), 256, 0, stream>>>(
            Xhi, Xlo, Pthi, Ptlo, xsq, psq, out);
    } else {
        xsq_kernel<<<BATCH / 8, 256, 0, stream>>>(X, xsq);
        psq_kernel<<<OUTDIM / 256, 256, 0, stream>>>(P, psq);
        proto_gemm_fused<<<dim3(BATCH / 128, OUTDIM / 128), 256, 0, stream>>>(
            X, P, xsq, psq, out);
    }
}
